// ExpertTier_60808146977375
// MI455X (gfx1250) — matrix operand order repaired and checked by reading, not compiled, not run on hardware
//
#include <hip/hip_runtime.h>
#include <hip/hip_bf16.h>
#include <math.h>

#define N_TOK   8192
#define DMODEL  1024
#define HHID    2048
#define NEXP    8
#define BM      64
#define BN      64
#define BK      32
#define LDA     40                    // halves per LDS row (32 + pad, 80B -> 16B aligned)
#define LDB     40
#define MAXPAD  (N_TOK + NEXP * BM)   // 8704 padded perm slots
#define MTILES  (MAXPAD / BM)         // 136 m-tiles

typedef __attribute__((ext_vector_type(4)))  float        f32x4;
typedef __attribute__((ext_vector_type(2)))  unsigned int u32x2;
typedef __attribute__((ext_vector_type(4)))  unsigned int u32x4;
typedef __attribute__((ext_vector_type(16))) __bf16       v16bf;
typedef __attribute__((ext_vector_type(8)))  float        v8f;

union FragBF { u32x4 u[2]; v16bf bf; };

__device__ __forceinline__ unsigned short f2bf(float f) {
  unsigned u = __float_as_uint(f);
  u += 0x7FFFu + ((u >> 16) & 1u);           // round-to-nearest-even
  return (unsigned short)(u >> 16);
}
__device__ __forceinline__ unsigned pack2(float a, float b) {
  return (unsigned)f2bf(a) | ((unsigned)f2bf(b) << 16);
}

// LDS byte offset of a generic shared pointer (ISA 10.2: low 32 bits of the address)
__device__ __forceinline__ unsigned lds_off(const void* p) {
  return (unsigned)(uintptr_t)p;
}

// CDNA5 async global->LDS copy, 16B per lane, tracked by ASYNCcnt (ISA 15.18.3 op 98)
__device__ __forceinline__ void async_copy16(unsigned ldsDst, const void* gsrc) {
  asm volatile("global_load_async_to_lds_b128 %0, %1, off"
               :: "v"(ldsDst), "v"((unsigned long long)(uintptr_t)gsrc)
               : "memory");
}
template <int N>
__device__ __forceinline__ void wait_asynccnt() {
  asm volatile("s_wait_asynccnt %0" :: "i"(N) : "memory");
}

// ---------------- one-time precision conversion (off the GEMM critical path) ----------------

// x (N,D) f32 -> bf16, same layout
__global__ __launch_bounds__(256) void cvt_x_kernel(const float* __restrict__ x,
                                                    unsigned short* __restrict__ xbf) {
  const int i = (blockIdx.x * 256 + threadIdx.x) * 4;
  f32x4 v = *(const f32x4*)(x + i);
  u32x2 p = { pack2(v[0], v[1]), pack2(v[2], v[3]) };
  *(u32x2*)(xbf + i) = p;
}

// src [E][R][C] f32 -> dst [E][C][R] bf16  (tiled 32x32 transpose via LDS)
__global__ __launch_bounds__(256) void transpose_cvt_kernel(const float* __restrict__ src,
                                                            unsigned short* __restrict__ dst,
                                                            int R, int C) {
  __shared__ unsigned short t[32][33];
  const size_t eOff = (size_t)blockIdx.z * R * C;
  const int rBase = blockIdx.y * 32, cBase = blockIdx.x * 32;
  const int tid = threadIdx.x;
  const int r = tid >> 3, c4 = (tid & 7) << 2;
  f32x4 v = *(const f32x4*)(src + eOff + (size_t)(rBase + r) * C + cBase + c4);
#pragma unroll
  for (int j = 0; j < 4; ++j) t[c4 + j][r] = f2bf(v[j]);
  __syncthreads();
  u32x2 o = { (unsigned)t[r][c4]     | ((unsigned)t[r][c4 + 1] << 16),
              (unsigned)t[r][c4 + 2] | ((unsigned)t[r][c4 + 3] << 16) };
  *(u32x2*)(dst + eOff + (size_t)(cBase + r) * R + rBase + c4) = o;
}

// ---------------- routing ----------------

__global__ void init_kernel(int* __restrict__ perm, int* __restrict__ count) {
  int i = blockIdx.x * blockDim.x + threadIdx.x;
  if (i < MAXPAD) perm[i] = -1;
  if (i < NEXP)  count[i] = 0;
}

__global__ void gate_kernel(const float* __restrict__ x, const float* __restrict__ gw,
                            float* __restrict__ topval, int* __restrict__ topidx,
                            int* __restrict__ count) {
  const int lane = threadIdx.x & 31;
  const int t = blockIdx.x * 8 + (threadIdx.x >> 5);   // one wave32 per token
  float acc[NEXP];
#pragma unroll
  for (int e = 0; e < NEXP; ++e) acc[e] = 0.f;
  const float* xr = x + (size_t)t * DMODEL;
  for (int j = lane; j < DMODEL; j += 32) {
    float xv = xr[j];
#pragma unroll
    for (int e = 0; e < NEXP; ++e) acc[e] += xv * gw[e * DMODEL + j];
  }
#pragma unroll
  for (int off = 16; off > 0; off >>= 1) {
#pragma unroll
    for (int e = 0; e < NEXP; ++e) acc[e] += __shfl_xor(acc[e], off, 32);
  }
  if (lane == 0) {
    float mx = acc[0]; int mi = 0;
#pragma unroll
    for (int e = 1; e < NEXP; ++e) if (acc[e] > mx) { mx = acc[e]; mi = e; }
    float s = 0.f;
#pragma unroll
    for (int e = 0; e < NEXP; ++e) s += __expf(acc[e] - mx);
    float p = 1.0f / s;                                  // top softmax prob
    topval[t] = p / (p + 1e-8f);
    topidx[t] = mi;
    atomicAdd(&count[mi], 1);
  }
}

__global__ void offsets_kernel(const int* __restrict__ count, int* __restrict__ off,
                               int* __restrict__ cursor) {
  if (threadIdx.x == 0) {
    int a = 0;
    off[0] = 0;
    for (int e = 0; e < NEXP; ++e) {
      a += (count[e] + BM - 1) & ~(BM - 1);              // pad segment to BM
      off[e + 1] = a;
      cursor[e] = 0;
    }
  }
}

__global__ void scatter_kernel(const int* __restrict__ topidx, const int* __restrict__ off,
                               int* __restrict__ cursor, int* __restrict__ perm) {
  int t = blockIdx.x * blockDim.x + threadIdx.x;
  if (t < N_TOK) {
    int e = topidx[t];
    int pos = off[e] + atomicAdd(&cursor[e], 1);
    perm[pos] = t;
  }
}

// ---------------- GEMM1: h = gelu(gather(xbf) @ W1[e]), bf16 WMMA ----------------
// Async double-buffered LDS staging; A: gathered bf16 rows; B: w1t[e][n][k].

__global__ __launch_bounds__(256) void gemm1_kernel(
    const unsigned short* __restrict__ xbf, const unsigned short* __restrict__ w1t,
    const int* __restrict__ perm, const int* __restrict__ off,
    unsigned short* __restrict__ h) {
  __shared__ unsigned short lA[2][BM * LDA];
  __shared__ unsigned short lB[2][BN * LDB];
  __shared__ int s_tok[BM];

  const int rowBase = blockIdx.y * BM;
  if (rowBase >= off[NEXP]) return;
  int e = 0;
  while (rowBase >= off[e + 1]) ++e;                     // tile lies within one expert segment
  const int nBase = blockIdx.x * BN;
  const int tid = threadIdx.x;
  if (tid < BM) s_tok[tid] = perm[rowBase + tid];
  __syncthreads();

  const int lane = tid & 31, wave = tid >> 5;
  const int nSub = wave & 3, mPair = wave >> 2;          // each wave: 2 m-subtiles x 1 n-subtile
  v8f c0 = {0.f,0.f,0.f,0.f,0.f,0.f,0.f,0.f};
  v8f c1 = {0.f,0.f,0.f,0.f,0.f,0.f,0.f,0.f};

  const int ar = tid >> 2, akg = (tid & 3) << 3;         // stage: 64 rows x 8 halves/thread
  const int bn = tid >> 2, bkg = (tid & 3) << 3;
  const int atok = s_tok[ar];
  const int atokc = (atok >= 0) ? atok : 0;              // pad rows: harmless dummy token
  const unsigned short* aSrc = xbf + (size_t)atokc * DMODEL + akg;
  const unsigned short* bSrc = w1t + (size_t)e * DMODEL * HHID
                                   + (size_t)(nBase + bn) * DMODEL + bkg;
  const unsigned aDst[2] = { lds_off(&lA[0][ar * LDA + akg]), lds_off(&lA[1][ar * LDA + akg]) };
  const unsigned bDst[2] = { lds_off(&lB[0][bn * LDB + bkg]), lds_off(&lB[1][bn * LDB + bkg]) };

  const int rowA0 = mPair * 32 + (lane & 15);
  const int akb   = (lane >> 4) << 3;                    // A frag: split K runs (ISA layout)
  const int brow  = nSub * 16 + (lane & 15);
  const int bkb   = (lane >> 4) << 3;                    // B frag: split K runs (ISA layout)

  // prologue: stage k=0 into buffer 0
  async_copy16(aDst[0], aSrc);
  async_copy16(bDst[0], bSrc);

  const int KT = DMODEL / BK;
  for (int k = 0; k < KT; ++k) {
    const int p = k & 1;
    __syncthreads();                                     // buf 1-p free to overwrite
    if (k + 1 < KT) {
      async_copy16(aDst[1 - p], aSrc + (k + 1) * BK);
      async_copy16(bDst[1 - p], bSrc + (k + 1) * BK);
      if (k + 2 < KT) __builtin_prefetch(bSrc + (k + 2) * BK, 0, 3);
      wait_asynccnt<2>();                                // stage k done, k+1 in flight
    } else {
      wait_asynccnt<0>();
    }
    __syncthreads();                                     // buf p visible to all waves

    FragBF fa0, fa1, fb;
    fa0.u[0] = *(const u32x4*)&lA[p][rowA0 * LDA + akb];
    fa0.u[1] = *(const u32x4*)&lA[p][rowA0 * LDA + akb + 16];
    fa1.u[0] = *(const u32x4*)&lA[p][(rowA0 + 16) * LDA + akb];
    fa1.u[1] = *(const u32x4*)&lA[p][(rowA0 + 16) * LDA + akb + 16];
    fb.u[0]  = *(const u32x4*)&lB[p][brow * LDB + bkb];
    fb.u[1]  = *(const u32x4*)&lB[p][brow * LDB + bkb + 16];
    c0 = __builtin_amdgcn_wmma_f32_16x16x32_bf16(false, fa0.bf, false, fb.bf,
                                                 (short)0, c0, false, false);
    c1 = __builtin_amdgcn_wmma_f32_16x16x32_bf16(false, fa1.bf, false, fb.bf,
                                                 (short)0, c1, false, false);
  }

  // exact-erf GELU, store bf16 h at padded slot rows
  const int col = nBase + nSub * 16 + (lane & 15);
#pragma unroll
  for (int i = 0; i < 8; ++i) {
    const int r0 = rowBase + mPair * 32 + i + ((lane >> 4) << 3);
    float v = c0[i];
    v = 0.5f * v * (1.0f + erff(v * 0.70710678118654752f));
    h[(size_t)r0 * HHID + col] = f2bf(v);
    float w = c1[i];
    w = 0.5f * w * (1.0f + erff(w * 0.70710678118654752f));
    h[(size_t)(r0 + 16) * HHID + col] = f2bf(w);
  }
}

// ---------------- GEMM2: out[token] = (h @ W2[e]) * top_val ----------------

__global__ __launch_bounds__(256) void gemm2_kernel(
    const unsigned short* __restrict__ h, const unsigned short* __restrict__ w2t,
    const int* __restrict__ perm, const int* __restrict__ off,
    const float* __restrict__ topval, float* __restrict__ out) {
  __shared__ unsigned short lA[2][BM * LDA];
  __shared__ unsigned short lB[2][BN * LDB];
  __shared__ int   s_tok[BM];
  __shared__ float s_val[BM];

  const int rowBase = blockIdx.y * BM;
  if (rowBase >= off[NEXP]) return;
  int e = 0;
  while (rowBase >= off[e + 1]) ++e;
  const int nBase = blockIdx.x * BN;
  const int tid = threadIdx.x;
  if (tid < BM) {
    int t = perm[rowBase + tid];
    s_tok[tid] = t;
    s_val[tid] = (t >= 0) ? topval[t] : 0.f;
  }
  __syncthreads();

  const int lane = tid & 31, wave = tid >> 5;
  const int nSub = wave & 3, mPair = wave >> 2;
  v8f c0 = {0.f,0.f,0.f,0.f,0.f,0.f,0.f,0.f};
  v8f c1 = {0.f,0.f,0.f,0.f,0.f,0.f,0.f,0.f};

  const int ar = tid >> 2, akg = (tid & 3) << 3;
  const int bn = tid >> 2, bkg = (tid & 3) << 3;
  const unsigned short* aSrc = h + (size_t)(rowBase + ar) * HHID + akg;
  const unsigned short* bSrc = w2t + (size_t)e * HHID * DMODEL
                                   + (size_t)(nBase + bn) * HHID + bkg;
  const unsigned aDst[2] = { lds_off(&lA[0][ar * LDA + akg]), lds_off(&lA[1][ar * LDA + akg]) };
  const unsigned bDst[2] = { lds_off(&lB[0][bn * LDB + bkg]), lds_off(&lB[1][bn * LDB + bkg]) };

  const int rowA0 = mPair * 32 + (lane & 15);
  const int akb   = (lane >> 4) << 3;
  const int brow  = nSub * 16 + (lane & 15);
  const int bkb   = (lane >> 4) << 3;

  async_copy16(aDst[0], aSrc);
  async_copy16(bDst[0], bSrc);

  const int KT = HHID / BK;
  for (int k = 0; k < KT; ++k) {
    const int p = k & 1;
    __syncthreads();
    if (k + 1 < KT) {
      async_copy16(aDst[1 - p], aSrc + (k + 1) * BK);
      async_copy16(bDst[1 - p], bSrc + (k + 1) * BK);
      if (k + 2 < KT) __builtin_prefetch(bSrc + (k + 2) * BK, 0, 3);
      wait_asynccnt<2>();
    } else {
      wait_asynccnt<0>();
    }
    __syncthreads();

    FragBF fa0, fa1, fb;
    fa0.u[0] = *(const u32x4*)&lA[p][rowA0 * LDA + akb];
    fa0.u[1] = *(const u32x4*)&lA[p][rowA0 * LDA + akb + 16];
    fa1.u[0] = *(const u32x4*)&lA[p][(rowA0 + 16) * LDA + akb];
    fa1.u[1] = *(const u32x4*)&lA[p][(rowA0 + 16) * LDA + akb + 16];
    fb.u[0]  = *(const u32x4*)&lB[p][brow * LDB + bkb];
    fb.u[1]  = *(const u32x4*)&lB[p][brow * LDB + bkb + 16];
    c0 = __builtin_amdgcn_wmma_f32_16x16x32_bf16(false, fa0.bf, false, fb.bf,
                                                 (short)0, c0, false, false);
    c1 = __builtin_amdgcn_wmma_f32_16x16x32_bf16(false, fa1.bf, false, fb.bf,
                                                 (short)0, c1, false, false);
  }

  // scale by top_val and scatter rows back to token order
  const int col = nBase + nSub * 16 + (lane & 15);
#pragma unroll
  for (int i = 0; i < 8; ++i) {
    const int m0 = mPair * 32 + i + ((lane >> 4) << 3);
    int t = s_tok[m0];
    if (t >= 0) out[(size_t)t * DMODEL + col] = c0[i] * s_val[m0];
    t = s_tok[m0 + 16];
    if (t >= 0) out[(size_t)t * DMODEL + col] = c1[i] * s_val[m0 + 16];
  }
}

// ---------------- LayerNorm (in place, one block per row) ----------------

__global__ __launch_bounds__(256) void ln_kernel(float* __restrict__ out,
                                                 const float* __restrict__ gamma,
                                                 const float* __restrict__ beta) {
  __shared__ float r1[256];
  __shared__ float r2[256];
  const int tid = threadIdx.x;
  float* row = out + (size_t)blockIdx.x * DMODEL;
  f32x4 v = *(const f32x4*)(row + tid * 4);
  r1[tid] = v[0] + v[1] + v[2] + v[3];
  r2[tid] = v[0]*v[0] + v[1]*v[1] + v[2]*v[2] + v[3]*v[3];
  __syncthreads();
  for (int o = 128; o > 0; o >>= 1) {
    if (tid < o) { r1[tid] += r1[tid + o]; r2[tid] += r2[tid + o]; }
    __syncthreads();
  }
  const float mu  = r1[0] * (1.0f / DMODEL);
  const float var = r2[0] * (1.0f / DMODEL) - mu * mu;
  const float rs  = rsqrtf(var + 1e-5f);
#pragma unroll
  for (int j = 0; j < 4; ++j) {
    const int cix = tid * 4 + j;
    row[cix] = (v[j] - mu) * rs * gamma[cix] + beta[cix];
  }
}

// ---------------- launcher ----------------

extern "C" void kernel_launch(void* const* d_in, const int* in_sizes, int n_in,
                              void* d_out, int out_size, void* d_ws, size_t ws_size,
                              hipStream_t stream) {
  const float* x     = (const float*)d_in[0];
  const float* gw    = (const float*)d_in[1];
  const float* w1    = (const float*)d_in[2];
  const float* w2    = (const float*)d_in[3];
  const float* gamma = (const float*)d_in[4];
  const float* beta  = (const float*)d_in[5];
  float* out = (float*)d_out;

  char* ws = (char*)d_ws;
  float* topval = (float*)ws;                         // 8192 f32
  int*   topidx = (int*)(ws + 32768);                 // 8192 i32
  int*   count  = (int*)(ws + 65536);                 // 8
  int*   cursor = (int*)(ws + 65536 + 32);            // 8
  int*   off    = (int*)(ws + 65536 + 64);            // 9
  int*   perm   = (int*)(ws + 65536 + 128);           // 8704 ints

  const size_t XBF_OFF = (size_t)1 << 17;
  const size_t W1T_OFF = XBF_OFF + (size_t)N_TOK * DMODEL * 2;            // +16MB
  const size_t W2T_OFF = W1T_OFF + (size_t)NEXP * DMODEL * HHID * 2;      // +32MB
  const size_t H_OFF   = W2T_OFF + (size_t)NEXP * HHID * DMODEL * 2;      // +32MB
  unsigned short* xbf = (unsigned short*)(ws + XBF_OFF);
  unsigned short* w1t = (unsigned short*)(ws + W1T_OFF);   // [E][H][D] bf16
  unsigned short* w2t = (unsigned short*)(ws + W2T_OFF);   // [E][D][H] bf16
  unsigned short* hbf = (unsigned short*)(ws + H_OFF);     // [MAXPAD][H] bf16

  // one-time precision conversion / weight transpose
  cvt_x_kernel<<<(N_TOK * DMODEL) / 1024, 256, 0, stream>>>(x, xbf);
  transpose_cvt_kernel<<<dim3(HHID / 32, DMODEL / 32, NEXP), 256, 0, stream>>>(w1, w1t, DMODEL, HHID);
  transpose_cvt_kernel<<<dim3(DMODEL / 32, HHID / 32, NEXP), 256, 0, stream>>>(w2, w2t, HHID, DMODEL);

  // routing
  init_kernel   <<<(MAXPAD + 255) / 256, 256, 0, stream>>>(perm, count);
  gate_kernel   <<<N_TOK / 8, 256, 0, stream>>>(x, gw, topval, topidx, count);
  offsets_kernel<<<1, 32, 0, stream>>>(count, off, cursor);
  scatter_kernel<<<N_TOK / 256, 256, 0, stream>>>(topidx, off, cursor, perm);

  // expert FFN (grouped bf16 WMMA GEMMs) + LayerNorm
  gemm1_kernel  <<<dim3(HHID / BN, MTILES), 256, 0, stream>>>(xbf, w1t, perm, off, hbf);
  gemm2_kernel  <<<dim3(DMODEL / BN, MTILES), 256, 0, stream>>>(hbf, w2t, perm, off, topval, out);
  ln_kernel     <<<N_TOK, 256, 0, stream>>>(out, gamma, beta);
}
